// DCLMBlock_16329465660009
// MI455X (gfx1250) — matrix operand order repaired and checked by reading, not compiled, not run on hardware
//
#include <hip/hip_runtime.h>

// ---------------- problem constants ----------------
#define B_      4
#define S_      2048
#define H_      1536
#define T_      (B_*S_)          // 8192 tokens
#define NH_     12
#define HD_     128
#define MEMD_   128
#define INNER_  6144
#define KW_     4
#define STRIDE_ 64
#define NC_     (S_/STRIDE_)     // 32 chunks

// ---------------- types ----------------
typedef __bf16 bf16_t;
typedef bf16_t v16bf __attribute__((ext_vector_type(16)));
typedef bf16_t v8bf  __attribute__((ext_vector_type(8)));
typedef bf16_t v4bf  __attribute__((ext_vector_type(4)));
typedef float  v8f   __attribute__((ext_vector_type(8)));

#define LDST  40            // LDS row stride in bf16 (80B: 16B-aligned chunks, bank-spread)
#define TILE_ (128*LDST)    // one 128x32 bf16 tile in LDS

__device__ __forceinline__ float sigmoidf_(float x) { return 1.0f / (1.0f + expf(-x)); }
__device__ __forceinline__ float geluf_(float x) {
  return 0.5f * x * (1.0f + erff(x * 0.70710678118654752f));
}

// ---------------- WMMA GEMM helpers ----------------
// Phase 1: issue all global loads for a 128x32 f32 tile into registers (no waits interleaved).
__device__ __forceinline__ void gload(const float* __restrict__ gp, long long ld,
                                      int k0, int tid, float4* regs) {
#pragma unroll
  for (int it = 0; it < 4; ++it) {
    int id  = tid + 256 * it;       // 0..1023
    int row = id >> 3;              // 0..127
    int c4  = (id & 7) << 2;        // 0,4,...,28
    regs[it] = *(const float4*)(gp + (long long)row * ld + k0 + c4);
  }
}
// Phase 2: convert f32 -> bf16 and store into LDS tile.
__device__ __forceinline__ void lstore(const float4* regs, bf16_t* lds, int tid) {
#pragma unroll
  for (int it = 0; it < 4; ++it) {
    int id  = tid + 256 * it;
    int row = id >> 3;
    int c4  = (id & 7) << 2;
    v4bf d;
    d[0] = (bf16_t)regs[it].x; d[1] = (bf16_t)regs[it].y;
    d[2] = (bf16_t)regs[it].z; d[3] = (bf16_t)regs[it].w;
    *(v4bf*)(&lds[row * LDST + c4]) = d;
  }
}
// Prefetch a tile two K-steps ahead (one branch per K-step at the call site).
__device__ __forceinline__ void gprefetch(const float* __restrict__ gp, long long ld,
                                          int k0, int tid) {
#pragma unroll
  for (int it = 0; it < 4; ++it) {
    int id  = tid + 256 * it;
    int row = id >> 3;
    int c4  = (id & 7) << 2;
    __builtin_prefetch(gp + (long long)row * ld + k0 + c4, 0, 1);
  }
}

// A fragment (16x32 bf16): lanes 0-15 hold K={0..7,16..23}, lanes 16-31 K={8..15,24..31}
__device__ __forceinline__ v16bf frag_a(const bf16_t* lds, int row0, int lane) {
  int r = row0 + (lane & 15);
  int g = lane >> 4;
  v8bf c0 = *(const v8bf*)(&lds[r * LDST + 8 * g]);
  v8bf c1 = *(const v8bf*)(&lds[r * LDST + 16 + 8 * g]);
  return __builtin_shufflevector(c0, c1, 0,1,2,3,4,5,6,7,8,9,10,11,12,13,14,15);
}
// B fragment (32x16 bf16): lane = column, lanes 0-15 hold K={0..7,16..23}, lanes 16-31 K={8..15,24..31}
__device__ __forceinline__ v16bf frag_b(const bf16_t* lds, int row0, int lane) {
  int r = row0 + (lane & 15);
  int g = lane >> 4;
  v8bf c0 = *(const v8bf*)(&lds[r * LDST + 8 * g]);
  v8bf c1 = *(const v8bf*)(&lds[r * LDST + 16 + 8 * g]);
  return __builtin_shufflevector(c0, c1, 0,1,2,3,4,5,6,7,8,9,10,11,12,13,14,15);
}

// ---------------- generic batched GEMM:  C = A * W^T (+bias) ----------------
// A: [M,K] row-major (lda), W: [N,K] row-major (ldw), C: [M,N] (ldc); batch via z strides.
// Grid: (N/128, M/128, batch). 256 threads. Requires K % 32 == 0, M,N % 128 == 0.
// Double-buffered LDS pipeline: one barrier per K-step, global loads overlap WMMA.
__global__ __launch_bounds__(256) void gemm_bias_kernel(
    const float* __restrict__ A, long long lda, long long sAz,
    const float* __restrict__ W, long long ldw, long long sWz,
    const float* __restrict__ bias,
    float* __restrict__ C, long long ldc, long long sCz, int K) {
  __shared__ bf16_t lA[2 * TILE_];
  __shared__ bf16_t lB[2 * TILE_];
  const float* Ab = A + (long long)blockIdx.z * sAz + (long long)blockIdx.y * 128 * lda;
  const float* Wb = W + (long long)blockIdx.z * sWz + (long long)blockIdx.x * 128 * ldw;
  float* Cb = C + (long long)blockIdx.z * sCz;
  const int tid = threadIdx.x, lane = tid & 31, wid = tid >> 5;
  const int wm = wid >> 2, wn = wid & 3;   // wave tile: rows wm*64.., cols wn*32..

  v8f acc[4][2] = {};
  float4 ra[4], rw[4];
  gload(Ab, lda, 0, tid, ra);
  gload(Wb, ldw, 0, tid, rw);
  int buf = 0;
  for (int k0 = 0; k0 < K; k0 += 32) {
    bf16_t* cA = lA + buf * TILE_;
    bf16_t* cB = lB + buf * TILE_;
    lstore(ra, cA, tid);
    lstore(rw, cB, tid);
    __syncthreads();
    if (k0 + 32 < K) {                       // issue next tile while WMMAs run
      gload(Ab, lda, k0 + 32, tid, ra);
      gload(Wb, ldw, k0 + 32, tid, rw);
    }
    if (k0 + 64 < K) {                       // L2 prefetch two tiles ahead
      gprefetch(Ab, lda, k0 + 64, tid);
      gprefetch(Wb, ldw, k0 + 64, tid);
    }
    v16bf bf[2];
#pragma unroll
    for (int nt = 0; nt < 2; ++nt) bf[nt] = frag_b(cB, wn * 32 + nt * 16, lane);
#pragma unroll
    for (int mt = 0; mt < 4; ++mt) {
      v16bf af = frag_a(cA, wm * 64 + mt * 16, lane);
#pragma unroll
      for (int nt = 0; nt < 2; ++nt)
        acc[mt][nt] = __builtin_amdgcn_wmma_f32_16x16x32_bf16(
            false, af, false, bf[nt], (short)0, acc[mt][nt], false, false);
    }
    buf ^= 1;
  }
  const int g = lane >> 4, nl = lane & 15;
#pragma unroll
  for (int mt = 0; mt < 4; ++mt)
#pragma unroll
    for (int nt = 0; nt < 2; ++nt) {
      int col = blockIdx.x * 128 + wn * 32 + nt * 16 + nl;
      float bv = bias ? bias[col] : 0.0f;
      long long rowb = (long long)blockIdx.y * 128 + wm * 64 + mt * 16 + 8 * g;
#pragma unroll
      for (int vr = 0; vr < 8; ++vr)
        Cb[(rowb + vr) * ldc + col] = acc[mt][nt][vr] + bv;
    }
}

// ---------------- GLU GEMM:  C = (A*W1^T) * sigmoid(A*W2^T) ----------------
// W1 = W rows [n..], W2 = W + wOff (wOff = Noff*ldw). Grid: (N/128, M/128).
__global__ __launch_bounds__(256) void gemm_glu_kernel(
    const float* __restrict__ A, long long lda,
    const float* __restrict__ W, long long ldw, long long wOff,
    float* __restrict__ C, long long ldc, int K) {
  __shared__ bf16_t lA[2 * TILE_];
  __shared__ bf16_t lB1[2 * TILE_];
  __shared__ bf16_t lB2[2 * TILE_];
  const float* Ab = A + (long long)blockIdx.y * 128 * lda;
  const float* W1 = W + (long long)blockIdx.x * 128 * ldw;
  const float* W2 = W + wOff + (long long)blockIdx.x * 128 * ldw;
  const int tid = threadIdx.x, lane = tid & 31, wid = tid >> 5;
  const int wm = wid >> 2, wn = wid & 3;

  v8f accC[4][2] = {};
  v8f accG[4][2] = {};
  float4 ra[4], rw1[4], rw2[4];
  gload(Ab, lda, 0, tid, ra);
  gload(W1, ldw, 0, tid, rw1);
  gload(W2, ldw, 0, tid, rw2);
  int buf = 0;
  for (int k0 = 0; k0 < K; k0 += 32) {
    bf16_t* cA  = lA  + buf * TILE_;
    bf16_t* cB1 = lB1 + buf * TILE_;
    bf16_t* cB2 = lB2 + buf * TILE_;
    lstore(ra,  cA,  tid);
    lstore(rw1, cB1, tid);
    lstore(rw2, cB2, tid);
    __syncthreads();
    if (k0 + 32 < K) {
      gload(Ab, lda, k0 + 32, tid, ra);
      gload(W1, ldw, k0 + 32, tid, rw1);
      gload(W2, ldw, k0 + 32, tid, rw2);
    }
    if (k0 + 64 < K) {
      gprefetch(Ab, lda, k0 + 64, tid);
      gprefetch(W1, ldw, k0 + 64, tid);
      gprefetch(W2, ldw, k0 + 64, tid);
    }
    v16bf b1[2], b2[2];
#pragma unroll
    for (int nt = 0; nt < 2; ++nt) {
      b1[nt] = frag_b(cB1, wn * 32 + nt * 16, lane);
      b2[nt] = frag_b(cB2, wn * 32 + nt * 16, lane);
    }
#pragma unroll
    for (int mt = 0; mt < 4; ++mt) {
      v16bf af = frag_a(cA, wm * 64 + mt * 16, lane);
#pragma unroll
      for (int nt = 0; nt < 2; ++nt) {
        accC[mt][nt] = __builtin_amdgcn_wmma_f32_16x16x32_bf16(
            false, af, false, b1[nt], (short)0, accC[mt][nt], false, false);
        accG[mt][nt] = __builtin_amdgcn_wmma_f32_16x16x32_bf16(
            false, af, false, b2[nt], (short)0, accG[mt][nt], false, false);
      }
    }
    buf ^= 1;
  }
  const int g = lane >> 4, nl = lane & 15;
#pragma unroll
  for (int mt = 0; mt < 4; ++mt)
#pragma unroll
    for (int nt = 0; nt < 2; ++nt) {
      int col = blockIdx.x * 128 + wn * 32 + nt * 16 + nl;
      long long rowb = (long long)blockIdx.y * 128 + wm * 64 + mt * 16 + 8 * g;
#pragma unroll
      for (int vr = 0; vr < 8; ++vr)
        C[(rowb + vr) * ldc + col] = accC[mt][nt][vr] * sigmoidf_(accG[mt][nt][vr]);
    }
}

// ---------------- elementwise / reduction kernels ----------------
__global__ void k_rms(const float* __restrict__ x, const float* __restrict__ w,
                      float* __restrict__ y) {
  long long t = blockIdx.x;
  const float* xr = x + t * H_;
  __shared__ float red[256];
  float s = 0.0f;
  for (int i = threadIdx.x; i < H_; i += 256) { float v = xr[i]; s += v * v; }
  red[threadIdx.x] = s; __syncthreads();
  for (int off = 128; off > 0; off >>= 1) {
    if (threadIdx.x < off) red[threadIdx.x] += red[threadIdx.x + off];
    __syncthreads();
  }
  float scale = rsqrtf(red[0] * (1.0f / H_) + 1e-6f);
  for (int i = threadIdx.x; i < H_; i += 256) y[t * H_ + i] = xr[i] * scale * w[i];
}

__global__ void k_token_gate(const float* __restrict__ x, const float* __restrict__ w,
                             const float* __restrict__ b, float* __restrict__ gate) {
  long long t = blockIdx.x;
  __shared__ float red[256];
  float s = 0.0f;
  for (int i = threadIdx.x; i < H_; i += 256) s += x[t * H_ + i] * w[i];
  red[threadIdx.x] = s; __syncthreads();
  for (int off = 128; off > 0; off >>= 1) {
    if (threadIdx.x < off) red[threadIdx.x] += red[threadIdx.x + off];
    __syncthreads();
  }
  if (threadIdx.x == 0) gate[t] = sigmoidf_(red[0] + b[0]);
}

__global__ void k_stack_conv(const float* __restrict__ in, float* __restrict__ out,
                             const float* __restrict__ w, const float* __restrict__ bias,
                             int dil) {
  long long gid = (long long)blockIdx.x * blockDim.x + threadIdx.x;
  if (gid >= (long long)T_ * H_) return;
  int c = (int)(gid % H_);
  long long t = gid / H_;
  int s = (int)(t % S_);
  float acc = bias[c];
#pragma unroll
  for (int kk = 0; kk < KW_; ++kk) {
    int ss = s - (KW_ - 1 - kk) * dil;
    if (ss >= 0) acc += w[c * KW_ + kk] * in[gid + (long long)(ss - s) * H_];
  }
  out[gid] = in[gid] + geluf_(acc);
}

__constant__ int cHeadDils[NH_][3] = {
  {1,2,4},{1,1,1},{4,8,16},{8,16,32},{32,64,128},{64,128,256},
  {256,512,1024},{1,100,200},{1,500,1000},{1,1024,2048},{3,9,27},{5,25,125}};

__global__ void k_head_conv(const float* __restrict__ in, float* __restrict__ out,
                            const float* __restrict__ w, const float* __restrict__ bias,
                            int j) {
  long long gid = (long long)blockIdx.x * blockDim.x + threadIdx.x;
  if (gid >= (long long)T_ * H_) return;
  int c = (int)(gid % H_);
  long long t = gid / H_;
  int s = (int)(t % S_);
  int i = c >> 7, hd = c & 127;
  int dil = cHeadDils[i][j];
  const float* wr = w + ((long long)(i * 3 + j) * HD_ + hd) * KW_;
  float acc = bias[(i * 3 + j) * HD_ + hd];
#pragma unroll
  for (int kk = 0; kk < KW_; ++kk) {
    int ss = s - (KW_ - 1 - kk) * dil;
    if (ss >= 0) acc += wr[kk] * in[gid + (long long)(ss - s) * H_];
  }
  out[gid] = in[gid] + acc;
}

__global__ void k_residual_gate(float* __restrict__ x, const float* __restrict__ y,
                                const float* __restrict__ gate) {
  long long gid = (long long)blockIdx.x * blockDim.x + threadIdx.x;
  if (gid >= (long long)T_ * H_) return;
  long long t = gid / H_;
  x[gid] += gate[t] * y[gid];
}

__global__ void k_mul_sigmoid(float* __restrict__ y, const float* __restrict__ z) {
  long long gid = (long long)blockIdx.x * blockDim.x + threadIdx.x;
  if (gid >= (long long)T_ * H_) return;
  y[gid] = y[gid] * sigmoidf_(z[gid]);
}

__global__ void k_head_router(const float* __restrict__ x, const float* __restrict__ w,
                              const float* __restrict__ b, float* __restrict__ hw) {
  long long t = blockIdx.x;
  __shared__ float accs[NH_];
  if (threadIdx.x < NH_) accs[threadIdx.x] = 0.0f;
  __syncthreads();
  float p[NH_];
#pragma unroll
  for (int i = 0; i < NH_; ++i) p[i] = 0.0f;
  for (int c = threadIdx.x; c < H_; c += 256) {
    float xv = x[t * H_ + c];
#pragma unroll
    for (int i = 0; i < NH_; ++i) p[i] += xv * w[i * H_ + c];
  }
#pragma unroll
  for (int i = 0; i < NH_; ++i) atomicAdd(&accs[i], p[i]);
  __syncthreads();
  if (threadIdx.x < NH_) hw[t * NH_ + threadIdx.x] = sigmoidf_(accs[threadIdx.x] + b[threadIdx.x]);
}

__global__ void k_mem_gate(const float* __restrict__ xh, const float* __restrict__ wg,
                           const float* __restrict__ bg, float* __restrict__ g) {
  long long idx = (long long)blockIdx.x * blockDim.x + threadIdx.x;
  if (idx >= (long long)4 * T_) return;
  int hh = (int)(idx / T_);
  long long t = idx % T_;
  const float* xr = xh + t * H_ + (6 + hh) * HD_;
  const float* wr = wg + hh * HD_;
  float s = 0.0f;
  for (int d = 0; d < HD_; ++d) s += xr[d] * wr[d];
  g[idx] = sigmoidf_(s + bg[hh]);
}

// Chunked linear-memory scan: one block per (head,batch); M state [128,128] in LDS.
__global__ __launch_bounds__(256) void k_scan(
    const float* __restrict__ q, const float* __restrict__ k,
    const float* __restrict__ v, const float* __restrict__ g,
    float* __restrict__ reads) {
  __shared__ float Msh[HD_ * MEMD_];   // 64 KB
  const int hh = blockIdx.x >> 2, b = blockIdx.x & 3;
  const long long base = (long long)hh * T_ * HD_ + (long long)b * S_ * HD_;
  const float* qb = q + base;
  const float* kb = k + base;
  const float* vb = v + (long long)hh * T_ * MEMD_ + (long long)b * S_ * MEMD_;
  const float* gb = g + (long long)hh * T_ + (long long)b * S_;
  float* rb = reads + (long long)hh * T_ * MEMD_ + (long long)b * S_ * MEMD_;
  for (int i = threadIdx.x; i < HD_ * MEMD_; i += 256) Msh[i] = 0.0f;
  __syncthreads();
  for (int c = 0; c < NC_; ++c) {
    const float* qc = qb + (long long)c * STRIDE_ * HD_;
    const float* kc = kb + (long long)c * STRIDE_ * HD_;
    const float* vc = vb + (long long)c * STRIDE_ * MEMD_;
    const float* gc = gb + c * STRIDE_;
    float* rc = rb + (long long)c * STRIDE_ * MEMD_;
    // reads = qc x M  (pre-update M)
    for (int r = 0; r < 32; ++r) {
      int o = threadIdx.x + 256 * r;
      int s = o >> 7, m = o & 127;
      float acc = 0.0f;
      for (int d = 0; d < HD_; ++d) acc += qc[s * HD_ + d] * Msh[d * MEMD_ + m];
      rc[s * MEMD_ + m] = acc;
    }
    float gsum = 0.0f;
    for (int s = 0; s < STRIDE_; ++s) gsum += gc[s];
    float decay = 1.0f - gsum * (1.0f / STRIDE_);
    __syncthreads();
    // M = decay*M + (k*g)^T v
    for (int r = 0; r < 64; ++r) {
      int idx = threadIdx.x + 256 * r;
      int d = idx >> 7, m = idx & 127;
      float acc = Msh[idx] * decay;
      for (int s = 0; s < STRIDE_; ++s) acc += kc[s * HD_ + d] * gc[s] * vc[s * MEMD_ + m];
      Msh[idx] = acc;
    }
    __syncthreads();
  }
}

__global__ void k_state_combine(const float* __restrict__ h, const float* __restrict__ memout,
                                const float* __restrict__ hw, float* __restrict__ out) {
  long long gid = (long long)blockIdx.x * blockDim.x + threadIdx.x;
  if (gid >= (long long)T_ * H_) return;
  int c = (int)(gid % H_);
  long long t = gid / H_;
  int i = c >> 7, d = c & 127;
  float v = h[gid];
  if (i >= 6 && i <= 9) v += memout[(long long)(i - 6) * T_ * HD_ + t * HD_ + d];
  out[gid] = v * hw[t * NH_ + i];
}

// ---------------- launcher ----------------
extern "C" void kernel_launch(void* const* d_in, const int* in_sizes, int n_in,
                              void* d_out, int out_size, void* d_ws, size_t ws_size,
                              hipStream_t stream) {
  (void)in_sizes; (void)n_in; (void)out_size; (void)ws_size;
  const float* x            = (const float*)d_in[0];
  const float* norm1_w      = (const float*)d_in[1];
  const float* norm2_w      = (const float*)d_in[2];
  const float* norm3_w      = (const float*)d_in[3];
  const float* convstack_w  = (const float*)d_in[4];
  const float* convstack_b  = (const float*)d_in[5];
  const float* conv_proj_w  = (const float*)d_in[6];
  const float* conv_proj_b  = (const float*)d_in[7];
  const float* gate_proj_w  = (const float*)d_in[8];
  const float* head_router_w= (const float*)d_in[9];
  const float* head_router_b= (const float*)d_in[10];
  const float* head_conv_w  = (const float*)d_in[11];
  const float* head_conv_b  = (const float*)d_in[12];
  const float* mem_Wq       = (const float*)d_in[13];
  const float* mem_Wk       = (const float*)d_in[14];
  const float* mem_Wv       = (const float*)d_in[15];
  const float* mem_Wg_w     = (const float*)d_in[16];
  const float* mem_Wg_b     = (const float*)d_in[17];
  const float* mem_Wout     = (const float*)d_in[18];
  const float* mix_gate_w   = (const float*)d_in[19];
  const float* mix_gate_b   = (const float*)d_in[20];
  const float* mixing_w     = (const float*)d_in[21];
  const float* mixing_b     = (const float*)d_in[22];
  const float* ffn_in_w     = (const float*)d_in[23];
  const float* ffn_out_w    = (const float*)d_in[24];
  const float* conv_gate_w  = (const float*)d_in[25];
  const float* conv_gate_b  = (const float*)d_in[26];
  const float* state_gate_w = (const float*)d_in[27];
  const float* state_gate_b = (const float*)d_in[28];
  const float* ffn_gate_w   = (const float*)d_in[29];
  const float* ffn_gate_b   = (const float*)d_in[30];

  float* xcur = (float*)d_out;

  // workspace carve-out (f32)
  float* bufA  = (float*)d_ws;                                   // T*H
  float* bufB  = bufA + (long long)T_ * H_;                      // T*H
  float* qbuf  = bufB + (long long)T_ * H_;                      // 4*T*HD
  float* kbuf  = qbuf + (long long)4 * T_ * HD_;
  float* vbuf  = kbuf + (long long)4 * T_ * HD_;
  float* gmem  = vbuf + (long long)4 * T_ * HD_;                 // 4*T
  float* rbuf  = gmem + (long long)4 * T_;                       // 4*T*MEM
  float* mout  = rbuf + (long long)4 * T_ * MEMD_;               // 4*T*HD
  float* hwbuf = mout + (long long)4 * T_ * HD_;                 // T*NH
  float* gate1 = hwbuf + (long long)T_ * NH_;                    // T
  float* gate2 = gate1 + T_;
  float* gate3 = gate2 + T_;
  float* hbig  = gate3 + T_;                                     // T*INNER

  hipMemcpyAsync(xcur, x, sizeof(float) * (size_t)T_ * H_, hipMemcpyDeviceToDevice, stream);

  dim3 blk(256);
  dim3 gridTH((unsigned)(((long long)T_ * H_ + 255) / 256));

  // ---------------- stage 1: conv block ----------------
  k_rms<<<T_, 256, 0, stream>>>(xcur, norm1_w, bufA);
  const int stackDils[6] = {1, 2, 4, 8, 16, 32};
  float* pin = bufA; float* pout = bufB;
  for (int j = 0; j < 6; ++j) {
    k_stack_conv<<<gridTH, blk, 0, stream>>>(pin, pout,
        convstack_w + (long long)j * H_ * KW_, convstack_b + (long long)j * H_, stackDils[j]);
    float* tmp = pin; pin = pout; pout = tmp;
  }
  k_token_gate<<<T_, 256, 0, stream>>>(xcur, conv_gate_w, conv_gate_b, gate1);
  gemm_bias_kernel<<<dim3(H_ / 128, T_ / 128, 1), blk, 0, stream>>>(
      pin, H_, 0, conv_proj_w, H_, 0, conv_proj_b, pout, H_, 0, H_);
  k_residual_gate<<<gridTH, blk, 0, stream>>>(xcur, pout, gate1);

  // ---------------- stage 2: state block ----------------
  k_token_gate<<<T_, 256, 0, stream>>>(xcur, state_gate_w, state_gate_b, gate2);
  k_rms<<<T_, 256, 0, stream>>>(xcur, norm2_w, bufA);
  k_head_router<<<T_, 256, 0, stream>>>(bufA, head_router_w, head_router_b, hwbuf);
  // xh = a * sigmoid(g)  (gate_proj GLU)  -> bufB [T, NH*HD]
  gemm_glu_kernel<<<dim3(H_ / 128, T_ / 128, 1), blk, 0, stream>>>(
      bufA, H_, gate_proj_w, H_, (long long)H_ * H_, bufB, H_, H_);
  // memory projections (batched over 4 heads)
  gemm_bias_kernel<<<dim3(1, T_ / 128, 4), blk, 0, stream>>>(
      bufB + 6 * HD_, H_, HD_, mem_Wq, HD_, (long long)HD_ * HD_, nullptr,
      qbuf, HD_, (long long)T_ * HD_, HD_);
  gemm_bias_kernel<<<dim3(1, T_ / 128, 4), blk, 0, stream>>>(
      bufB + 6 * HD_, H_, HD_, mem_Wk, HD_, (long long)HD_ * HD_, nullptr,
      kbuf, HD_, (long long)T_ * HD_, HD_);
  gemm_bias_kernel<<<dim3(1, T_ / 128, 4), blk, 0, stream>>>(
      bufB + 6 * HD_, H_, HD_, mem_Wv, HD_, (long long)MEMD_ * HD_, nullptr,
      vbuf, MEMD_, (long long)T_ * MEMD_, HD_);
  k_mem_gate<<<(4 * T_ + 255) / 256, blk, 0, stream>>>(bufB, mem_Wg_w, mem_Wg_b, gmem);
  k_scan<<<16, 256, 0, stream>>>(qbuf, kbuf, vbuf, gmem, rbuf);
  gemm_bias_kernel<<<dim3(1, T_ / 128, 4), blk, 0, stream>>>(
      rbuf, MEMD_, (long long)T_ * MEMD_, mem_Wout, MEMD_, (long long)HD_ * MEMD_, nullptr,
      mout, HD_, (long long)T_ * HD_, MEMD_);
  // per-head dilated convs (3 sequential passes) on xh
  k_head_conv<<<gridTH, blk, 0, stream>>>(bufB, bufA, head_conv_w, head_conv_b, 0);
  k_head_conv<<<gridTH, blk, 0, stream>>>(bufA, bufB, head_conv_w, head_conv_b, 1);
  k_head_conv<<<gridTH, blk, 0, stream>>>(bufB, bufA, head_conv_w, head_conv_b, 2);
  k_state_combine<<<gridTH, blk, 0, stream>>>(bufA, mout, hwbuf, bufB);
  // out = out * sigmoid(out @ mix_gate_w^T + b)
  gemm_bias_kernel<<<dim3(H_ / 128, T_ / 128, 1), blk, 0, stream>>>(
      bufB, H_, 0, mix_gate_w, H_, 0, mix_gate_b, bufA, H_, 0, H_);
  k_mul_sigmoid<<<gridTH, blk, 0, stream>>>(bufB, bufA);
  // mixing projection
  gemm_bias_kernel<<<dim3(H_ / 128, T_ / 128, 1), blk, 0, stream>>>(
      bufB, H_, 0, mixing_w, H_, 0, mixing_b, bufA, H_, 0, H_);
  k_residual_gate<<<gridTH, blk, 0, stream>>>(xcur, bufA, gate2);

  // ---------------- stage 3: GLU FFN ----------------
  k_token_gate<<<T_, 256, 0, stream>>>(xcur, ffn_gate_w, ffn_gate_b, gate3);
  k_rms<<<T_, 256, 0, stream>>>(xcur, norm3_w, bufA);
  gemm_glu_kernel<<<dim3(INNER_ / 128, T_ / 128, 1), blk, 0, stream>>>(
      bufA, H_, ffn_in_w, H_, (long long)INNER_ * H_, hbig, INNER_, H_);
  gemm_bias_kernel<<<dim3(H_ / 128, T_ / 128, 1), blk, 0, stream>>>(
      hbig, INNER_, 0, ffn_out_w, INNER_, 0, nullptr, bufB, H_, 0, INNER_);
  k_residual_gate<<<gridTH, blk, 0, stream>>>(xcur, bufB, gate3);
}
